// RGCN_30090540876234
// MI455X (gfx1250) — matrix operand order repaired and checked by reading, not compiled, not run on hardware
//
#include <hip/hip_runtime.h>

typedef __attribute__((ext_vector_type(16))) __bf16 v16bf;
typedef __attribute__((ext_vector_type(8)))  float  v8f;

union ABu {
    v16bf v;
    uint4 q[2];
};

__device__ __forceinline__ unsigned short f32_to_bf16_rne(float f) {
    unsigned int u = __float_as_uint(f);
    unsigned int r = 0x7FFFu + ((u >> 16) & 1u);
    u += r;
    return (unsigned short)(u >> 16);
}

__device__ __forceinline__ float bf16_to_f32(unsigned short h) {
    return __uint_as_float(((unsigned int)h) << 16);
}

// Transpose + hi/lo bf16 split of the 9 weight matrices (8 bases + self-loop).
// Input  V:[8][128][128] (k-major), loopw:[128][128] (k-major)
// Output wt:[9][128][128] laid out [mat][n][k] so B-fragments are contiguous.
__global__ __launch_bounds__(256) void prep_wt(
        const float* __restrict__ V,
        const float* __restrict__ loopw,
        unsigned short* __restrict__ wt_hi,
        unsigned short* __restrict__ wt_lo)
{
    int i = blockIdx.x * blockDim.x + threadIdx.x;
    if (i >= 9 * 128 * 128) return;
    int mat = i >> 14;
    int rem = i & 16383;
    int n = rem >> 7;
    int k = rem & 127;
    float w = (mat < 8) ? V[(mat * 128 + k) * 128 + n] : loopw[k * 128 + n];
    unsigned short h = f32_to_bf16_rne(w);
    wt_hi[i] = h;
    wt_lo[i] = f32_to_bf16_rne(w - bf16_to_f32(h));
}

// f32 -> (hi, lo) bf16 split, optional ReLU (used to feed layer-2 GEMM).
__global__ __launch_bounds__(256) void split_bf16(
        const float* __restrict__ in,
        unsigned short* __restrict__ hi,
        unsigned short* __restrict__ lo,
        int n, int relu)
{
    int i = blockIdx.x * blockDim.x + threadIdx.x;
    if (i >= n) return;
    float x = in[i];
    if (relu) x = fmaxf(x, 0.0f);
    unsigned short h = f32_to_bf16_rne(x);
    hi[i] = h;
    lo[i] = f32_to_bf16_rne(x - bf16_to_f32(h));
}

// Error-compensated bf16 WMMA GEMM:
//   mat < 8 : XB[m][mat][0:128] = h[m] @ V[mat]     (f32 accum)
//   mat == 8: agg[m][0:128]     = h[m] @ loop + bias
// One wave computes a 16x128 strip; block = 8 waves -> 128 rows.
__global__ __launch_bounds__(256) void rgcn_gemm(
        const unsigned short* __restrict__ h_hi,   // [N][128] bf16
        const unsigned short* __restrict__ h_lo,   // [N][128] bf16
        const unsigned short* __restrict__ wt_hi,  // [9][128][128] ([mat][n][k])
        const unsigned short* __restrict__ wt_lo,
        const float* __restrict__ bias,            // [128], used when mat==8
        float* __restrict__ xb,                    // [N][8][128]
        float* __restrict__ agg,                   // [N][128]
        int N)
{
    const int mat  = blockIdx.y;
    const int wave = threadIdx.x >> 5;
    const int lane = threadIdx.x & 31;
    const int lh   = lane >> 4;    // half-wave select
    const int l16  = lane & 15;
    const int m0   = blockIdx.x * 128 + wave * 16;

    // A fragment source row (clamped so EXEC stays all-ones for WMMA)
    int ar = m0 + l16;
    if (ar > N - 1) ar = N - 1;
    const unsigned short* ah = h_hi + (size_t)ar * 128 + lh * 16;
    const unsigned short* al = h_lo + (size_t)ar * 128 + lh * 16;
    // B fragment base: column n = nt*16 + l16 of W^T, 16 consecutive k per half-wave
    const unsigned short* wh = wt_hi + ((size_t)mat * 128 + l16) * 128 + lh * 16;
    const unsigned short* wl = wt_lo + ((size_t)mat * 128 + l16) * 128 + lh * 16;

    v8f acc[8];
    const v8f zero = {0.f, 0.f, 0.f, 0.f, 0.f, 0.f, 0.f, 0.f};
#pragma unroll
    for (int t = 0; t < 8; ++t) acc[t] = zero;

#pragma unroll
    for (int ks = 0; ks < 4; ++ks) {
        // A 16x32 bf16 fragment: lane row = l16, 16 consecutive K {off..off+15}
        ABu a_hi, a_lo;
        {
            const uint4* p = (const uint4*)(ah + ks * 32);
            a_hi.q[0] = p[0]; a_hi.q[1] = p[1];
            const uint4* q = (const uint4*)(al + ks * 32);
            a_lo.q[0] = q[0]; a_lo.q[1] = q[1];
        }
#pragma unroll
        for (int nt = 0; nt < 8; ++nt) {
            // B 32x16 bf16 fragment: 16 consecutive k of column nt*16+l16
            ABu b_hi, b_lo;
            const uint4* p = (const uint4*)(wh + (size_t)nt * 16 * 128 + ks * 32);
            b_hi.q[0] = p[0]; b_hi.q[1] = p[1];
            const uint4* q = (const uint4*)(wl + (size_t)nt * 16 * 128 + ks * 32);
            b_lo.q[0] = q[0]; b_lo.q[1] = q[1];

            acc[nt] = __builtin_amdgcn_wmma_f32_16x16x32_bf16(
                false, a_hi.v, false, b_hi.v, (short)0, acc[nt], false, false);
            acc[nt] = __builtin_amdgcn_wmma_f32_16x16x32_bf16(
                false, a_lo.v, false, b_hi.v, (short)0, acc[nt], false, false);
            acc[nt] = __builtin_amdgcn_wmma_f32_16x16x32_bf16(
                false, a_hi.v, false, b_lo.v, (short)0, acc[nt], false, false);
        }
    }

    // D layout: VGPR i -> row m0 + i + 8*lh, col = nt*16 + l16
#pragma unroll
    for (int nt = 0; nt < 8; ++nt) {
        const int n = nt * 16 + l16;
#pragma unroll
        for (int i = 0; i < 8; ++i) {
            const int m = m0 + i + lh * 8;
            if (m < N) {
                if (mat < 8) xb[((size_t)m * 8 + mat) * 128 + n] = acc[nt][i];
                else         agg[(size_t)m * 128 + n] = acc[nt][i] + bias[n];
            }
        }
    }
}

// One wave per edge: contiguous 4KB gather of XB[src], combine with
// comp[etype] coefficients and norm, atomic scatter-add into agg[dst].
__global__ __launch_bounds__(256) void rgcn_edges(
        const float* __restrict__ xb,     // [N][8][128]
        const int* __restrict__ src,
        const int* __restrict__ dst,
        const int* __restrict__ etype,
        const float* __restrict__ norm,   // [E]
        const float* __restrict__ comp,   // [R][8]
        float* __restrict__ agg,          // [N][128]
        int E)
{
    const int wave = threadIdx.x >> 5;
    const int lane = threadIdx.x & 31;
    const int e = blockIdx.x * 8 + wave;
    if (e >= E) return;

    const int s = src[e];
    const int d = dst[e];
    const int r = etype[e];
    const float nm = norm[e];

    float c[8];
#pragma unroll
    for (int b = 0; b < 8; ++b) c[b] = comp[r * 8 + b];

    const float* xp = xb + (size_t)s * 1024 + lane * 4;
    float a0 = 0.f, a1 = 0.f, a2 = 0.f, a3 = 0.f;
#pragma unroll
    for (int b = 0; b < 8; ++b) {
        float4 x = *(const float4*)(xp + b * 128);
        a0 += c[b] * x.x;
        a1 += c[b] * x.y;
        a2 += c[b] * x.z;
        a3 += c[b] * x.w;
    }
    float* o = agg + (size_t)d * 128 + lane * 4;
    atomicAdd(o + 0, nm * a0);
    atomicAdd(o + 1, nm * a1);
    atomicAdd(o + 2, nm * a2);
    atomicAdd(o + 3, nm * a3);
}

extern "C" void kernel_launch(void* const* d_in, const int* in_sizes, int n_in,
                              void* d_out, int out_size, void* d_ws, size_t ws_size,
                              hipStream_t stream)
{
    const float* feats = (const float*)d_in[0];
    const int*   src   = (const int*)d_in[1];
    const int*   dst   = (const int*)d_in[2];
    const int*   etype = (const int*)d_in[3];
    const float* norm  = (const float*)d_in[4];
    const float* V1    = (const float*)d_in[5];
    const float* comp1 = (const float*)d_in[6];
    const float* loop1 = (const float*)d_in[7];
    const float* bias1 = (const float*)d_in[8];
    const float* V2    = (const float*)d_in[9];
    const float* comp2 = (const float*)d_in[10];
    const float* loop2 = (const float*)d_in[11];
    const float* bias2 = (const float*)d_in[12];

    const int N = in_sizes[0] / 128;
    const int E = in_sizes[1];

    char* ws = (char*)d_ws;
    size_t off = 0;
    auto take = [&](size_t bytes) -> char* {
        char* p = ws + off;
        off += (bytes + 255) & ~(size_t)255;
        return p;
    };
    float*          XB   = (float*)take((size_t)N * 8 * 128 * sizeof(float));
    float*          AGG1 = (float*)take((size_t)N * 128 * sizeof(float));
    unsigned short* Hhi  = (unsigned short*)take((size_t)N * 128 * 2);
    unsigned short* Hlo  = (unsigned short*)take((size_t)N * 128 * 2);
    unsigned short* W1h  = (unsigned short*)take((size_t)9 * 128 * 128 * 2);
    unsigned short* W1l  = (unsigned short*)take((size_t)9 * 128 * 128 * 2);
    unsigned short* W2h  = (unsigned short*)take((size_t)9 * 128 * 128 * 2);
    unsigned short* W2l  = (unsigned short*)take((size_t)9 * 128 * 128 * 2);

    const int wblocks = (9 * 128 * 128 + 255) / 256;
    prep_wt<<<wblocks, 256, 0, stream>>>(V1, loop1, W1h, W1l);
    prep_wt<<<wblocks, 256, 0, stream>>>(V2, loop2, W2h, W2l);

    const int nh = N * 128;
    const int sblocks = (nh + 255) / 256;
    split_bf16<<<sblocks, 256, 0, stream>>>(feats, Hhi, Hlo, nh, 0);

    dim3 gg((N + 127) / 128, 9);

    // ---- layer 1 ----
    rgcn_gemm<<<gg, 256, 0, stream>>>(Hhi, Hlo, W1h, W1l, bias1, XB, AGG1, N);
    rgcn_edges<<<(E + 7) / 8, 256, 0, stream>>>(XB, src, dst, etype, norm, comp1, AGG1, E);

    // ReLU + re-split for layer 2
    split_bf16<<<sblocks, 256, 0, stream>>>(AGG1, Hhi, Hlo, nh, 1);

    // ---- layer 2 (writes final output; mat==8 epilogue fully initializes d_out) ----
    rgcn_gemm<<<gg, 256, 0, stream>>>(Hhi, Hlo, W2h, W2l, bias2, XB, (float*)d_out, N);
    rgcn_edges<<<(E + 7) / 8, 256, 0, stream>>>(XB, src, dst, etype, norm, comp2, (float*)d_out, E);
}
